// CausalSelfAttention_42717744726423
// MI455X (gfx1250) — compile-verified, matrix operand order checked by reading, not run on hardware
//
#include <hip/hip_runtime.h>
#include <hip/hip_bf16.h>

#define DMODEL 2048
#define NHEAD  16
#define HDIM   128
#define SEQ    2048
#define BATCH  4

typedef __bf16 bf16;
typedef __bf16 v16bf __attribute__((ext_vector_type(16)));
typedef __bf16 v8bf  __attribute__((ext_vector_type(8)));
typedef __bf16 v4bf  __attribute__((ext_vector_type(4)));
typedef float  v8f   __attribute__((ext_vector_type(8)));
typedef unsigned int u32x4 __attribute__((ext_vector_type(4)));
typedef int          i32x4 __attribute__((ext_vector_type(4)));
typedef int          i32x8 __attribute__((ext_vector_type(8)));

union FragU { v16bf v; v8bf h[2]; };

static __device__ __forceinline__ v16bf load_frag(const bf16* p0, const bf16* p1) {
  FragU f;
  f.h[0] = *(const v8bf*)p0;
  f.h[1] = *(const v8bf*)p1;
  return f.v;
}

static __device__ __forceinline__ v8f wmma_bf16(v16bf a, v16bf b, v8f c) {
  return __builtin_amdgcn_wmma_f32_16x16x32_bf16(false, a, false, b, (short)0, c, false, false);
}

// ---------------------------------------------------------------------------
// Tensor Data Mover: 2D bf16 tile (tile_d1 rows x tile_d0 elems) -> LDS.
// D# built per CDNA5 ISA (group0: count/lds/global addr,
// group1: data_size=2B, dims/strides).  Issued by one wave; completion via
// TENSORcnt (s_wait_tensorcnt).  6-arg builtin form (clang-23 toolchain).
// ---------------------------------------------------------------------------
static __device__ __forceinline__ void tdm_load_2d(
    unsigned lds_addr, const void* gptr,
    unsigned tensor_d0, unsigned tensor_d1,
    unsigned tile_d0, unsigned tile_d1,
    unsigned long long stride_elems)
{
  unsigned long long ga = (unsigned long long)gptr;
  u32x4 g0;
  g0[0] = 1u;                                               // count = 1 (valid)
  g0[1] = lds_addr;                                         // LDS byte address
  g0[2] = (unsigned)(ga & 0xffffffffu);                     // global_addr[31:0]
  g0[3] = (unsigned)((ga >> 32) & 0x1ffffffu) | (2u << 30); // addr[56:32] | type=2
  i32x8 g1;
  g1[0] = (int)(1u << 16);                                  // data_size = 1 -> 2B
  g1[1] = (int)((tensor_d0 & 0xffffu) << 16);               // tensor_dim0[15:0]
  g1[2] = (int)((tensor_d0 >> 16) | ((tensor_d1 & 0xffffu) << 16));
  g1[3] = (int)(((tensor_d1 >> 16) & 0xffffu) | (tile_d0 << 16));
  g1[4] = (int)(tile_d1 & 0xffffu);                         // tile_dim1 (tile_dim2=0)
  g1[5] = (int)(stride_elems & 0xffffffffull);              // tensor_dim0_stride lo
  g1[6] = (int)((stride_elems >> 32) & 0xffffull);          // stride hi | dim1_stride=0
  g1[7] = 0;
  i32x4 z4 = {0, 0, 0, 0};                                  // groups 2/3 unused (2D)
  i32x8 z8 = {0, 0, 0, 0, 0, 0, 0, 0};
  __builtin_amdgcn_tensor_load_to_lds(g0, g1, z4, z4, z8, 0);
}

// ---------------------------------------------------------------------------
// fp32 -> bf16 conversion, 4-wide
// ---------------------------------------------------------------------------
__global__ void cvt_f32_to_bf16_v4(const float* __restrict__ in,
                                   bf16* __restrict__ out, int n4) {
  int i = blockIdx.x * blockDim.x + threadIdx.x;
  if (i < n4) {
    float4 f = ((const float4*)in)[i];
    v4bf o;
    o[0] = (bf16)f.x; o[1] = (bf16)f.y; o[2] = (bf16)f.z; o[3] = (bf16)f.w;
    ((v4bf*)out)[i] = o;
  }
}

// ---------------------------------------------------------------------------
// Tiled bf16 WMMA GEMM:  C[M,Ncols] = A[M,K] * B[K,Ncols] + bias
//   mode 0: fp32 output to Cout
//   mode 1: bf16 scatter into Q/K/V head layout [B,H,N,HD]
// BM=128 BN=128 BK=32, 256 threads (8 waves), wave grid 4(m) x 2(n).
// Double-buffered LDS: A tiles via TDM (issued by wave 0, overlapped one
// iteration ahead), B tiles via register-staged cooperative loads.
// ---------------------------------------------------------------------------
#define BM 128
#define BN 128
#define BK 32

__global__ __launch_bounds__(256)
void gemm_bf16_wmma(const bf16* __restrict__ A, const bf16* __restrict__ Bm,
                    const float* __restrict__ bias,
                    float* __restrict__ Cout,
                    bf16* __restrict__ Qd, bf16* __restrict__ Kd, bf16* __restrict__ Vd,
                    int M, int Ncols, int Kdim, int mode)
{
  __shared__ __align__(16) bf16 As[2][BM][BK];    // [buf][m][k]
  __shared__ __align__(16) bf16 BsT[2][BN][BK];   // [buf][n][k]

  const unsigned tid  = threadIdx.x;
  const unsigned lane = tid & 31u;
  const unsigned wave = tid >> 5;
  const unsigned wm   = wave & 3u;     // 32-row strip
  const unsigned wn   = wave >> 2;     // 64-col strip
  const unsigned lr   = lane & 15u;
  const unsigned hs   = lane >> 4;
  const unsigned m0   = blockIdx.x * BM;
  const unsigned n0   = blockIdx.y * BN;

  const unsigned ldsAs0 = (unsigned)(unsigned long long)(void*)&As[0][0][0];
  const unsigned ldsAs1 = (unsigned)(unsigned long long)(void*)&As[1][0][0];

  v8f zero = {};
  v8f acc[2][4];
#pragma unroll
  for (int i = 0; i < 2; i++)
#pragma unroll
    for (int j = 0; j < 4; j++) acc[i][j] = zero;

  // B-tile staging (global -> regs -> LDS transposed)
  const unsigned kk   = tid >> 3;             // 0..31
  const unsigned nseg = (tid & 7u) * 16;      // 0..112
  v8bf b0, b1;
  auto gloadB = [&](int kt) {
    const bf16* src = Bm + (size_t)((unsigned)kt * BK + kk) * (unsigned)Ncols + n0 + nseg;
    b0 = *(const v8bf*)src;
    b1 = *(const v8bf*)(src + 8);
  };
  auto storeB = [&](int buf) {
#pragma unroll
    for (int e = 0; e < 8; e++) {
      BsT[buf][nseg + e][kk]     = b0[e];
      BsT[buf][nseg + 8 + e][kk] = b1[e];
    }
  };

  const int nk = Kdim / BK;
  // Prologue: tile 0 in flight
  if (wave == 0)
    tdm_load_2d(ldsAs0, A + (size_t)m0 * (unsigned)Kdim,
                (unsigned)Kdim, (unsigned)M, BK, BM, (unsigned long long)Kdim);
  gloadB(0);
  storeB(0);

  for (int k = 0; k < nk; ++k) {
    const int cur = k & 1, nxt = cur ^ 1;
    if (wave == 0) __builtin_amdgcn_s_wait_tensorcnt(0);  // A_k DMA complete
    __syncthreads();                                      // all tiles visible
    if (k + 1 < nk) {
      if (wave == 0)
        tdm_load_2d(nxt ? ldsAs1 : ldsAs0,
                    A + (size_t)m0 * (unsigned)Kdim + (size_t)(k + 1) * BK,
                    (unsigned)Kdim, (unsigned)M, BK, BM, (unsigned long long)Kdim);
      gloadB(k + 1);
    }

    v16bf afr[2], bfr[4];
#pragma unroll
    for (int i = 0; i < 2; i++) {
      const bf16* p = &As[cur][wm * 32 + i * 16 + lr][hs * 8];
      afr[i] = load_frag(p, p + 16);
    }
#pragma unroll
    for (int j = 0; j < 4; j++) {
      const bf16* p = &BsT[cur][wn * 64 + j * 16 + lr][hs * 8];
      bfr[j] = load_frag(p, p + 16);
    }
#pragma unroll
    for (int i = 0; i < 2; i++)
#pragma unroll
      for (int j = 0; j < 4; j++)
        acc[i][j] = wmma_bf16(afr[i], bfr[j], acc[i][j]);

    if (k + 1 < nk) storeB(nxt);
  }

  // Epilogue.  C/D layout: vgpr g, lane l -> row = g + (l/16)*8, col = l%16
#pragma unroll
  for (int i = 0; i < 2; i++) {
#pragma unroll
    for (int j = 0; j < 4; j++) {
#pragma unroll
      for (int g = 0; g < 8; g++) {
        unsigned r = m0 + wm * 32 + i * 16 + g + hs * 8;
        unsigned c = n0 + wn * 64 + j * 16 + lr;
        float val = acc[i][j][g] + bias[c];
        if (mode == 0) {
          Cout[(size_t)r * (unsigned)Ncols + c] = val;
        } else {
          unsigned which = c >> 11;          // / DMODEL
          unsigned dd    = c & 2047u;        // % DMODEL
          unsigned h  = dd >> 7, hd = dd & 127u;
          unsigned b  = r >> 11, t  = r & 2047u;
          bf16* dst = (which == 0) ? Qd : ((which == 1) ? Kd : Vd);
          dst[(((size_t)(b * NHEAD + h)) * SEQ + t) * HDIM + hd] = (bf16)val;
        }
      }
    }
  }
}

// ---------------------------------------------------------------------------
// Flash attention (causal, online softmax), bf16 WMMA, fp32 accumulate.
// Grid: (B*H, SEQ/64).  128 threads = 4 waves; wave w owns q rows [q0+16w,+15].
// K tile (32x128) fetched by the TDM; V tile transposed cooperatively into LDS.
// Uniform trip count per workgroup -> barriers safe under causal masking.
// ---------------------------------------------------------------------------
__global__ __launch_bounds__(128)
void flash_attn_wmma(const bf16* __restrict__ Q, const bf16* __restrict__ Kt,
                     const bf16* __restrict__ V, bf16* __restrict__ AO)
{
  __shared__ __align__(16) bf16 Ks[32][HDIM];    // [kv][hd]   (TDM dest)
  __shared__ __align__(16) bf16 VsT[HDIM][32];   // [hd][kv]   (transposed)
  __shared__ __align__(16) bf16 Ps[4][16][32];   // per-wave P tile

  const unsigned tid  = threadIdx.x;
  const unsigned lane = tid & 31u;
  const unsigned wave = tid >> 5;
  const unsigned lr   = lane & 15u;
  const unsigned hs   = lane >> 4;

  const unsigned bh = blockIdx.x;            // b*H + h
  const unsigned q0 = blockIdx.y * 64;       // q block start
  const unsigned qw = q0 + wave * 16;        // this wave's q tile start
  const size_t base = (size_t)bh * SEQ * HDIM;
  const unsigned ldsKs = (unsigned)(unsigned long long)(void*)&Ks[0][0];

  // Q fragments: 16x128 as 4 A-fragments (16x32 each)
  v16bf qa[4];
  {
    const bf16* qrow = Q + base + (size_t)(qw + lr) * HDIM;
#pragma unroll
    for (int c = 0; c < 4; c++) {
      const bf16* p = qrow + c * 32 + hs * 8;
      qa[c] = load_frag(p, p + 16);
    }
  }

  v8f zero = {};
  v8f o[8];
#pragma unroll
  for (int t = 0; t < 8; t++) o[t] = zero;
  float mrow[8], lrow[8];
#pragma unroll
  for (int g = 0; g < 8; g++) { mrow[g] = -3.0e38f; lrow[g] = 0.0f; }

  const float scale = 0.08838834764831845f;   // 1/sqrt(128)
  const int iters = (int)(q0 / 32 + 2);       // cover kv <= q0+63

  for (int it = 0; it < iters; ++it) {
    const unsigned kv0 = (unsigned)it * 32;
    __syncthreads();                      // previous tile fully consumed
    if (wave == 0)                        // K tile via Tensor Data Mover
      tdm_load_2d(ldsKs, Kt + base + (size_t)kv0 * HDIM,
                  HDIM, SEQ, HDIM, 32, HDIM);
    // V tile (32x128) -> VsT transposed, 32 elems/thread
    {
      unsigned row = tid >> 2;
      unsigned seg = (tid & 3u) * 32;
      const bf16* vsrc = V + base + (size_t)(kv0 + row) * HDIM + seg;
      v8bf v0 = *(const v8bf*)(vsrc);
      v8bf v1 = *(const v8bf*)(vsrc + 8);
      v8bf v2 = *(const v8bf*)(vsrc + 16);
      v8bf v3 = *(const v8bf*)(vsrc + 24);
#pragma unroll
      for (int e = 0; e < 8; e++) {
        VsT[seg + e][row]      = v0[e];
        VsT[seg + 8 + e][row]  = v1[e];
        VsT[seg + 16 + e][row] = v2[e];
        VsT[seg + 24 + e][row] = v3[e];
      }
    }
    if (wave == 0) __builtin_amdgcn_s_wait_tensorcnt(0);
    __syncthreads();

    // S = Q K^T : two 16x16 tiles over kv step of 32
    v8f s0 = zero, s1 = zero;
#pragma unroll
    for (int c = 0; c < 4; c++) {
      const bf16* p0 = &Ks[lr][c * 32 + hs * 8];
      const bf16* p1 = &Ks[16 + lr][c * 32 + hs * 8];
      v16bf kb0 = load_frag(p0, p0 + 16);
      v16bf kb1 = load_frag(p1, p1 + 16);
      s0 = wmma_bf16(qa[c], kb0, s0);
      s1 = wmma_bf16(qa[c], kb1, s1);
    }

    // Causal mask + online softmax (row r = g + hs*8, col = lr / lr+16)
#pragma unroll
    for (int g = 0; g < 8; g++) {
      unsigned qrow = qw + g + hs * 8;
      unsigned c0 = kv0 + lr;
      unsigned c1 = c0 + 16;
      float x0 = (c0 <= qrow) ? s0[g] * scale : -3.0e38f;
      float x1 = (c1 <= qrow) ? s1[g] * scale : -3.0e38f;
      float xm = fmaxf(x0, x1);
#pragma unroll
      for (int off = 1; off < 16; off <<= 1) xm = fmaxf(xm, __shfl_xor(xm, off));
      float mnew = fmaxf(mrow[g], xm);
      float corr = __expf(mrow[g] - mnew);
      float p0 = __expf(x0 - mnew);
      float p1 = __expf(x1 - mnew);
      float ps = p0 + p1;
#pragma unroll
      for (int off = 1; off < 16; off <<= 1) ps += __shfl_xor(ps, off);
      lrow[g] = lrow[g] * corr + ps;
      mrow[g] = mnew;
#pragma unroll
      for (int t = 0; t < 8; t++) o[t][g] *= corr;
      unsigned pr = g + hs * 8;
      Ps[wave][pr][lr]      = (bf16)p0;
      Ps[wave][pr][lr + 16] = (bf16)p1;
    }
    // per-wave DS ops are in-order; ensure writes land before A-frag reload
    asm volatile("s_wait_dscnt 0" ::: "memory");

    // O += P @ V
    {
      const bf16* pp = &Ps[wave][lr][hs * 8];
      v16bf pa = load_frag(pp, pp + 16);
#pragma unroll
      for (int t = 0; t < 8; t++) {
        const bf16* vp = &VsT[t * 16 + lr][hs * 8];
        v16bf vb = load_frag(vp, vp + 16);
        o[t] = wmma_bf16(pa, vb, o[t]);
      }
    }
  }

  // Normalize and store into [B, N, D] bf16 layout for the out-projection
  const unsigned b = bh >> 4, h = bh & 15u;   // / %, NHEAD = 16
#pragma unroll
  for (int g = 0; g < 8; g++) {
    float inv = 1.0f / lrow[g];
    unsigned qrow = qw + g + hs * 8;
    bf16* dst = AO + ((size_t)(b * SEQ + qrow)) * DMODEL + h * HDIM;
#pragma unroll
    for (int t = 0; t < 8; t++) dst[t * 16 + lr] = (bf16)(o[t][g] * inv);
  }
}

// ---------------------------------------------------------------------------
// launch
// ---------------------------------------------------------------------------
extern "C" void kernel_launch(void* const* d_in, const int* in_sizes, int n_in,
                              void* d_out, int out_size, void* d_ws, size_t ws_size,
                              hipStream_t stream)
{
  const float* x    = (const float*)d_in[0];
  const float* Wqkv = (const float*)d_in[1];
  const float* bqkv = (const float*)d_in[2];
  const float* Wout = (const float*)d_in[3];
  const float* bout = (const float*)d_in[4];
  float* out = (float*)d_out;

  char* ws = (char*)d_ws;
  size_t off = 0;
  auto alloc = [&](size_t bytes) {
    char* p = ws + off;
    off += (bytes + 255) & ~(size_t)255;
    return p;
  };
  const size_t ND  = (size_t)BATCH * SEQ * DMODEL;       // 16.8M elems
  bf16* xb  = (bf16*)alloc(ND * 2);
  bf16* wqb = (bf16*)alloc((size_t)DMODEL * 3 * DMODEL * 2);
  bf16* wob = (bf16*)alloc((size_t)DMODEL * DMODEL * 2);
  bf16* q   = (bf16*)alloc(ND * 2);
  bf16* k   = (bf16*)alloc(ND * 2);
  bf16* v   = (bf16*)alloc(ND * 2);
  bf16* ao  = (bf16*)alloc(ND * 2);

  auto cvt = [&](const float* src, bf16* dst, size_t n) {
    int n4 = (int)(n / 4);
    cvt_f32_to_bf16_v4<<<(n4 + 255) / 256, 256, 0, stream>>>(src, dst, n4);
  };
  cvt(x,    xb,  ND);
  cvt(Wqkv, wqb, (size_t)DMODEL * 3 * DMODEL);
  cvt(Wout, wob, (size_t)DMODEL * DMODEL);

  // QKV projection: M=8192, N=6144, K=2048  -> scatter into q/k/v heads
  {
    dim3 grid((BATCH * SEQ) / BM, (3 * DMODEL) / BN);
    gemm_bf16_wmma<<<grid, 256, 0, stream>>>(xb, wqb, bqkv, nullptr, q, k, v,
                                             BATCH * SEQ, 3 * DMODEL, DMODEL, 1);
  }
  // Causal flash attention
  {
    dim3 grid(BATCH * NHEAD, SEQ / 64);
    flash_attn_wmma<<<grid, 128, 0, stream>>>(q, k, v, ao);
  }
  // Output projection: M=8192, N=2048, K=2048 -> fp32 d_out
  {
    dim3 grid((BATCH * SEQ) / BM, DMODEL / BN);
    gemm_bf16_wmma<<<grid, 256, 0, stream>>>(ao, wob, bout, out,
                                             nullptr, nullptr, nullptr,
                                             BATCH * SEQ, DMODEL, DMODEL, 0);
  }
}
